// MoELayer_66279935312554
// MI455X (gfx1250) — matrix operand order repaired and checked by reading, not compiled, not run on hardware
//
#include <hip/hip_runtime.h>

// ---------------- problem constants (match reference) ----------------
#define NTOK  8192            // B*T
#define DIMD  1024
#define DIMH  2048
#define DIMO  1024
#define NEXP  8
#define NSLOT (2 * NTOK)      // top-2 assignments

// ---------------- tiling ----------------
#define TM   64               // rows (tokens/slots) per block tile
#define TN   128              // output cols per chunk
#define TKC  128              // K staging chunk
#define LDA  (TKC + 8)        // LDS row stride for A tiles (bf16 elems)

typedef __attribute__((ext_vector_type(16))) __bf16 v16bf;
typedef __attribute__((ext_vector_type(4)))  __bf16 v4bf;
typedef __attribute__((ext_vector_type(8)))  float  v8f;
typedef __attribute__((ext_vector_type(4)))  unsigned int u32x4;
typedef __attribute__((ext_vector_type(8)))  unsigned int u32x8;

// ---------------- workspace layout (bytes) ----------------
#define WS_CNT    0
#define WS_OFFS   32
#define WS_CURSOR 64
#define WS_E0     128
#define WS_E1     (WS_E0 + NTOK * 4)
#define WS_W0     (WS_E1 + NTOK * 4)
#define WS_W1G    (WS_W0 + NTOK * 4)
#define WS_TOK    (WS_W1G + NTOK * 4)
#define WS_WOF    (WS_TOK + NSLOT * 4)
#define WS_H      (WS_WOF + NSLOT * 4 + 128)   // 16-aligned, bf16 [NSLOT][DIMH]

// ============ WMMA fragment access (per CDNA5 ISA VGPR layouts) ============
// A: 16x32 bf16, row-major LDS tile with stride LDA.
//   lanes 0-15: row=lane, kh=0; lanes 16-31: row=lane-16, kh=8.
//   elems 0-7 = K kh..kh+7 (16B contig), elems 8-15 = K 16+kh.. (16B contig).
__device__ __forceinline__ v16bf frag_a(const __bf16* s, int rowBase, int kk, int lane) {
  const int r  = rowBase + (lane & 15);
  const int kh = (lane >> 4) << 3;
  const __bf16* p = s + r * LDA + kk + kh;
  v16bf a;
#pragma unroll
  for (int i = 0; i < 8; ++i) a[i] = p[i];
#pragma unroll
  for (int i = 0; i < 8; ++i) a[8 + i] = p[16 + i];
  return a;
}
// B: stored pre-swizzled in LDS as [kc][ct][lane][16 contiguous bf16], so a
// fragment load is one 32B contiguous read per lane (2x ds_load_b128).
__device__ __forceinline__ v16bf frag_b_swz(const __bf16* sB, int ct, int kc, int lane) {
  return *(const v16bf*)(sB + (((((kc << 3) + ct) << 5) + lane) << 4));
}
// Stage a TKC x TN f32 panel of W (row stride ldn) into swizzled bf16 LDS.
// Thread handles 4 "runs"; each run = 16 K at one column (two groups of 8, 16 apart) ->
// coalesced global reads across lanes, two ds_store_b128 per run.
__device__ __forceinline__ void stage_b_swz(const float* __restrict__ W, int ldn,
                                            __bf16* sB, int tid) {
#pragma unroll
  for (int rr = 0; rr < 4; ++rr) {
    const int r    = tid + (rr << 8);       // 0..1023
    const int c    = r & 127;
    const int krun = r >> 7;                // 0..7 (16 K each)
    const int kc   = krun >> 1;
    const int kh   = krun & 1;
    const int lane = (c & 15) + (kh << 4);
    const int ct   = c >> 4;
    const float* src = W + (size_t)((kc << 5) + (kh << 3)) * ldn + c;
    v16bf tmp;
#pragma unroll
    for (int j = 0; j < 16; ++j) tmp[j] = (__bf16)src[(size_t)(j + (j & 8)) * ldn];
    *(v16bf*)(sB + (((((kc << 3) + ct) << 5) + lane) << 4)) = tmp;
  }
}

// ============================ 1) gating ============================
__global__ __launch_bounds__(256) void moe_gate(
    const float* __restrict__ x, const float* __restrict__ gW,
    const float* __restrict__ gb, int* __restrict__ cnt,
    int* __restrict__ e0A, int* __restrict__ e1A,
    float* __restrict__ w0A, float* __restrict__ w1A)
{
  const int t    = (int)((blockIdx.x * blockDim.x + threadIdx.x) >> 5);
  const int lane = threadIdx.x & 31;
  if (t >= NTOK) return;
  const float* xr = x + (size_t)t * DIMD;
  float acc[NEXP];
#pragma unroll
  for (int e = 0; e < NEXP; ++e) acc[e] = 0.f;
  for (int d = lane; d < DIMD; d += 32) {
    const float xv = xr[d];
    const float* gr = gW + (size_t)d * NEXP;
#pragma unroll
    for (int e = 0; e < NEXP; ++e) acc[e] = fmaf(xv, gr[e], acc[e]);
  }
#pragma unroll
  for (int off = 16; off > 0; off >>= 1) {
#pragma unroll
    for (int e = 0; e < NEXP; ++e) acc[e] += __shfl_xor(acc[e], off, 32);
  }
  if (lane == 0) {
    float mx = -3.4e38f;
#pragma unroll
    for (int e = 0; e < NEXP; ++e) { acc[e] += gb[e]; mx = fmaxf(mx, acc[e]); }
    float p[NEXP];
#pragma unroll
    for (int e = 0; e < NEXP; ++e) p[e] = __expf(acc[e] - mx);
    int b0 = 0;
#pragma unroll
    for (int e = 1; e < NEXP; ++e) if (p[e] > p[b0]) b0 = e;     // ties -> lowest idx
    int b1 = (b0 == 0) ? 1 : 0;
#pragma unroll
    for (int e = 0; e < NEXP; ++e) if (e != b0 && p[e] > p[b1]) b1 = e;
    const float s = p[b0] + p[b1];           // renorm is softmax-scale invariant
    e0A[t] = b0; e1A[t] = b1;
    w0A[t] = p[b0] / s; w1A[t] = p[b1] / s;
    atomicAdd(&cnt[b0], 1);
    atomicAdd(&cnt[b1], 1);
  }
}

// ============================ 2) scan ============================
__global__ void moe_scan(const int* __restrict__ cnt, int* __restrict__ offs,
                         int* __restrict__ cursor)
{
  if (threadIdx.x == 0) {
    int a = 0;
#pragma unroll
    for (int e = 0; e < NEXP; ++e) { offs[e] = a; a += cnt[e]; cursor[e] = 0; }
  }
}

// ============================ 3) scatter ============================
__global__ __launch_bounds__(256) void moe_scatter(
    const int* __restrict__ e0A, const int* __restrict__ e1A,
    const float* __restrict__ w0A, const float* __restrict__ w1A,
    const int* __restrict__ offs, int* __restrict__ cursor,
    int* __restrict__ tokOf, float* __restrict__ wOf)
{
  const int t = blockIdx.x * blockDim.x + threadIdx.x;
  if (t >= NTOK) return;
  int e0 = e0A[t];
  int s0 = offs[e0] + atomicAdd(&cursor[e0], 1);
  tokOf[s0] = t; wOf[s0] = w0A[t];
  int e1 = e1A[t];
  int s1 = offs[e1] + atomicAdd(&cursor[e1], 1);
  tokOf[s1] = t; wOf[s1] = w1A[t];
}

// ==================== 4) grouped GEMM1: H = relu(Xg @ W1[e] + b1) ====================
__global__ __launch_bounds__(256) void moe_ffn1(
    const float* __restrict__ x, const float* __restrict__ W1,
    const float* __restrict__ b1, const int* __restrict__ cnt,
    const int* __restrict__ offs, const int* __restrict__ tokOf,
    __bf16* __restrict__ Hws)
{
  __shared__ alignas(32) __bf16 sA[TM * LDA];                  // 17 KB
  __shared__ alignas(32) __bf16 sB[(TKC / 32) * (TN / 16) * 32 * 16]; // 32 KB swizzled
  const int e  = blockIdx.y;
  const int n  = cnt[e];
  const int m0 = blockIdx.x * TM;
  if (m0 >= n) return;                       // block-uniform: EXEC stays all-ones
  const int base = offs[e];
  const float* We = W1 + (size_t)e * DIMD * DIMH;
  const float* be = b1 + (size_t)e * DIMH;
  const int tid  = threadIdx.x;
  const int lane = tid & 31;
  const int wv   = tid >> 5;
  const int wRow = (wv & 3) << 4;            // 0,16,32,48
  const int wCol = (wv >> 2) << 2;           // tile-col base: 0 or 4 (of 8 ct)

  // A staging: 4 threads per row, 32 floats each
  const int srow = tid >> 2, ssub = tid & 3;
  int grow = m0 + srow; if (grow >= n) grow = n - 1;   // clamp pad rows
  const int tok = tokOf[base + grow];
  const float* xrow = x + (size_t)tok * DIMD + ssub * 32;

  for (int nc = 0; nc < DIMH / TN; ++nc) {
    v8f acc[4];
#pragma unroll
    for (int ct = 0; ct < 4; ++ct)
#pragma unroll
      for (int i = 0; i < 8; ++i) acc[ct][i] = 0.f;

    for (int kb = 0; kb < DIMD; kb += TKC) {
      { // stage gathered X rows, f32 -> bf16, packed ds_store_b64
        const float4* src = (const float4*)(xrow + kb);
        __bf16* dst = sA + srow * LDA + ssub * 32;
#pragma unroll
        for (int i = 0; i < 8; ++i) {
          float4 v = src[i];
          v4bf pk; pk[0] = (__bf16)v.x; pk[1] = (__bf16)v.y;
          pk[2] = (__bf16)v.z; pk[3] = (__bf16)v.w;
          *(v4bf*)(dst + 4 * i) = pk;
        }
      }
      stage_b_swz(We + (size_t)kb * DIMH + nc * TN, DIMH, sB, tid);
      if (kb + TKC < DIMD)
        __builtin_prefetch(We + (size_t)(kb + TKC + (tid & 127)) * DIMH + nc * TN + ((tid >> 7) << 6), 0, 0);
      __syncthreads();
#pragma unroll
      for (int kc = 0; kc < TKC / 32; ++kc) {
        v16bf a = frag_a(sA, wRow, kc << 5, lane);
#pragma unroll
        for (int ct = 0; ct < 4; ++ct) {
          v16bf b = frag_b_swz(sB, wCol + ct, kc, lane);
          acc[ct] = __builtin_amdgcn_wmma_f32_16x16x32_bf16(
              false, a, false, b, (short)0, acc[ct], false, false);
        }
      }
      __syncthreads();
    }
    // epilogue: bias + relu, store bf16 H (C/D layout: VGPR i -> M = i + 8*(lane>>4))
    const int colG = nc * TN + (wCol << 4);
#pragma unroll
    for (int ct = 0; ct < 4; ++ct) {
      const int col = colG + (ct << 4) + (lane & 15);
      const float bia = be[col];
#pragma unroll
      for (int i = 0; i < 8; ++i) {
        const int row = wRow + i + ((lane >> 4) << 3);
        if (m0 + row < n) {
          float v = acc[ct][i] + bia;
          v = v > 0.f ? v : 0.f;
          Hws[(size_t)(base + m0 + row) * DIMH + col] = (__bf16)v;
        }
      }
    }
  }
}

// ============ 5) grouped GEMM2: out[tok] += w * (H @ W2[e] + b2) ============
__global__ __launch_bounds__(256) void moe_ffn2(
    const __bf16* __restrict__ Hws, const float* __restrict__ W2,
    const float* __restrict__ b2, const int* __restrict__ cnt,
    const int* __restrict__ offs, const int* __restrict__ tokOf,
    const float* __restrict__ wOf, float* __restrict__ out)
{
  __shared__ alignas(32) __bf16 sA[TM * LDA];
  __shared__ alignas(32) __bf16 sB[(TKC / 32) * (TN / 16) * 32 * 16];
  const int e  = blockIdx.y;
  const int n  = cnt[e];
  const int m0 = blockIdx.x * TM;
  if (m0 >= n) return;
  const int base = offs[e];
  const float* We = W2 + (size_t)e * DIMH * DIMO;
  const float* be = b2 + (size_t)e * DIMO;
  const int tid  = threadIdx.x;
  const int lane = tid & 31;
  const int wv   = tid >> 5;
  const int wRow = (wv & 3) << 4;
  const int wCol = (wv >> 2) << 2;

  const int rows = (n - m0 < TM) ? (n - m0) : TM;   // valid rows; TDM zero-fills rest
  const unsigned lds_base = (unsigned)(size_t)(void*)sA;  // flat LDS addr[31:0] = LDS offset

  for (int nc = 0; nc < DIMO / TN; ++nc) {
    v8f acc[4];
#pragma unroll
    for (int ct = 0; ct < 4; ++ct)
#pragma unroll
      for (int i = 0; i < 8; ++i) acc[ct][i] = 0.f;

    for (int kb = 0; kb < DIMH; kb += TKC) {
      // ---- Tensor Data Mover: async copy H panel [rows x 128] bf16 -> sA ----
      // D# (2D): data_size=2B, tile 128x64, tensor_dim1=rows (OOB rows read 0),
      // pad 4 DWORDs every 64 DWORDs -> LDS row stride LDA=136 bf16.
      if (wv == 0) {
        const unsigned long long ga =
            (unsigned long long)(size_t)(const void*)(Hws + (size_t)(base + m0) * DIMH + kb);
        u32x4 g0;
        g0[0] = 1u;                                           // count=1
        g0[1] = lds_base;                                     // lds_addr
        g0[2] = (unsigned)(ga & 0xffffffffu);                 // global_addr[31:0]
        g0[3] = (unsigned)((ga >> 32) & 0x1ffffffu) | (2u << 30); // addr[56:32] | type=2
        u32x8 g1;
        g1[0] = (1u << 16) | (1u << 20) | (5u << 22) | (3u << 25); // dsize=2B,pad 4DW/64DW
        g1[1] = (128u & 0xffffu) << 16;                       // tensor_dim0 = 128
        g1[2] = ((unsigned)rows & 0xffffu) << 16;             // tensor_dim1 = rows
        g1[3] = (128u << 16);                                 // tile_dim0 = 128
        g1[4] = 64u;                                          // tile_dim1 = 64
        g1[5] = (unsigned)DIMH;                               // tensor_dim0_stride = 2048
        g1[6] = 0u;
        g1[7] = 0u;
        asm volatile("tensor_load_to_lds %0, %1" :: "s"(g0), "s"(g1) : "memory");
      }
      // overlap: stage W2 K-panel (f32 -> bf16, swizzled) while TDM runs
      stage_b_swz(We + (size_t)kb * DIMO + nc * TN, DIMO, sB, tid);
      if (kb + TKC < DIMH)
        __builtin_prefetch(We + (size_t)(kb + TKC + (tid & 127)) * DIMO + nc * TN + ((tid >> 7) << 6), 0, 0);
      if (wv == 0) __builtin_amdgcn_s_wait_tensorcnt(0);
      __syncthreads();
#pragma unroll
      for (int kc = 0; kc < TKC / 32; ++kc) {
        v16bf a = frag_a(sA, wRow, kc << 5, lane);
#pragma unroll
        for (int ct = 0; ct < 4; ++ct) {
          v16bf b = frag_b_swz(sB, wCol + ct, kc, lane);
          acc[ct] = __builtin_amdgcn_wmma_f32_16x16x32_bf16(
              false, a, false, b, (short)0, acc[ct], false, false);
        }
      }
      __syncthreads();
    }
    // epilogue: +b2, scale by gate weight, atomic accumulate into out[token]
    const int colG = nc * TN + (wCol << 4);
#pragma unroll
    for (int ct = 0; ct < 4; ++ct) {
      const int col = colG + (ct << 4) + (lane & 15);
      const float bia = be[col];
#pragma unroll
      for (int i = 0; i < 8; ++i) {
        const int row = wRow + i + ((lane >> 4) << 3);
        if (m0 + row < n) {
          const int slot = base + m0 + row;
          const float v = (acc[ct][i] + bia) * wOf[slot];
          atomicAdd(out + (size_t)tokOf[slot] * DIMO + col, v);
        }
      }
    }
  }
}

// ============================ host launcher ============================
extern "C" void kernel_launch(void* const* d_in, const int* in_sizes, int n_in,
                              void* d_out, int out_size, void* d_ws, size_t ws_size,
                              hipStream_t stream) {
  const float* x   = (const float*)d_in[0];
  const float* gW  = (const float*)d_in[1];
  const float* gb  = (const float*)d_in[2];
  const float* W1  = (const float*)d_in[3];
  const float* b1  = (const float*)d_in[4];
  const float* W2  = (const float*)d_in[5];
  const float* b2  = (const float*)d_in[6];
  // d_in[7] = num_experts_per_tok (== 2, compile-time here)
  float* out = (float*)d_out;
  char*  ws  = (char*)d_ws;

  int*    cnt    = (int*)(ws + WS_CNT);
  int*    offs   = (int*)(ws + WS_OFFS);
  int*    cursor = (int*)(ws + WS_CURSOR);
  int*    e0A    = (int*)(ws + WS_E0);
  int*    e1A    = (int*)(ws + WS_E1);
  float*  w0A    = (float*)(ws + WS_W0);
  float*  w1A    = (float*)(ws + WS_W1G);
  int*    tokOf  = (int*)(ws + WS_TOK);
  float*  wOf    = (float*)(ws + WS_WOF);
  __bf16* Hws    = (__bf16*)(ws + WS_H);

  hipMemsetAsync(ws, 0, 128, stream);                              // zero counters
  hipMemsetAsync(d_out, 0, (size_t)out_size * sizeof(float), stream);

  moe_gate<<<(NTOK * 32) / 256, 256, 0, stream>>>(x, gW, gb, cnt, e0A, e1A, w0A, w1A);
  moe_scan<<<1, 32, 0, stream>>>(cnt, offs, cursor);
  moe_scatter<<<NTOK / 256, 256, 0, stream>>>(e0A, e1A, w0A, w1A, offs, cursor, tokOf, wOf);

  dim3 grid(NTOK / TM, NEXP);   // over-provisioned; blocks past cnt[e] exit
  moe_ffn1<<<grid, 256, 0, stream>>>(x, W1, b1, cnt, offs, tokOf, Hws);
  moe_ffn2<<<grid, 256, 0, stream>>>(Hws, W2, b2, cnt, offs, tokOf, wOf, out);
}
